// FeatureSelfAttention_21517786153302
// MI455X (gfx1250) — matrix operand order repaired and checked by reading, not compiled, not run on hardware
//
#include <hip/hip_runtime.h>

// ---------------------------------------------------------------------------
// FeatureSelfAttention for MI455X (gfx1250, wave32)
//   out  = (x@WqT) * sum((x@WkT)*(x@WvT), -1) * 1/sqrt(D) @ WoT + bo
//   attn = outer(q, k) / sqrt(D)   (256x768x768 f32, ~604 MB -> streaming NT stores)
// GEMMs use v_wmma_f32_16x16x32_bf16 (f32 accumulate), software-pipelined
// (double-buffered fragments) so loads overlap WMMA issue.
// ---------------------------------------------------------------------------

#define DDIM 768
#define BATCH 256

typedef __attribute__((ext_vector_type(16))) __bf16 v16bf;
typedef __attribute__((ext_vector_type(8)))  float  v8f;
typedef __attribute__((ext_vector_type(4)))  float  f32x4;

union BFrag { int4 q[2]; v16bf v; };   // 16 bf16 = 32 bytes

__device__ __forceinline__ unsigned short f2bf(float f) {
  unsigned int u = __float_as_uint(f);
  u += 0x7FFFu + ((u >> 16) & 1u);     // round-to-nearest-even
  return (unsigned short)(u >> 16);
}

// ---------------- f32 -> bf16 conversion (vectorized x4) -------------------
__global__ void cvt_bf16_kernel(const float* __restrict__ in,
                                unsigned short* __restrict__ out, int n4) {
  int i = blockIdx.x * blockDim.x + threadIdx.x;
  if (i < n4) {
    f32x4 f = reinterpret_cast<const f32x4*>(in)[i];
    ushort4 o;
    o.x = f2bf(f.x); o.y = f2bf(f.y); o.z = f2bf(f.z); o.w = f2bf(f.w);
    reinterpret_cast<ushort4*>(out)[i] = o;
  }
}

// ---- fragment loaders -----------------------------------------------------
// A 16x32 bf16 fragment: lanes 0-15 row=lane K{0..15}; lanes 16-31
// row=lane-16 K{16..31}.
__device__ __forceinline__ void load_afrag(BFrag& f,
                                           const unsigned short* __restrict__ arow,
                                           int k0, int lhalf) {
  f.q[0] = *reinterpret_cast<const int4*>(arow + k0 + lhalf * 16);
  f.q[1] = *reinterpret_cast<const int4*>(arow + k0 + 8 + lhalf * 16);
}
// B 32x16 fragment (B = W^T): lane n holds W[n][k0..k0+15] contiguous
// (lanes 16-31: +16). wbase already includes (row * DDIM + lhalf*16).
__device__ __forceinline__ void load_bfrag(BFrag& f,
                                           const unsigned short* __restrict__ wbase,
                                           int k0) {
  const int4* p = reinterpret_cast<const int4*>(wbase + k0);
  f.q[0] = p[0];
  f.q[1] = p[1];
}

// ---------------- one wave computes a 16(M) x 64(N) tile of A @ W^T --------
// A: M x DDIM row-major bf16.  W: N x DDIM row-major bf16 (B = W^T).
// C: row-major f32 ld=DDIM.  Optional bias[n].
// K-loop is double buffered: step 64, two statically-named fragment sets.
__device__ __forceinline__ void gemm_wave_16x64(
    const unsigned short* __restrict__ A,
    const unsigned short* __restrict__ W,
    float* __restrict__ C,
    const float* __restrict__ bias,
    int m0, int n0) {
  const int lane  = threadIdx.x & 31;
  const int lhalf = lane >> 4;     // 0: lanes 0-15, 1: lanes 16-31
  const int lrow  = lane & 15;

  v8f acc[4] = { v8f{}, v8f{}, v8f{}, v8f{} };

  const unsigned short* arow = A + (m0 + lrow) * DDIM;
  const unsigned short* wb[4];
#pragma unroll
  for (int nt = 0; nt < 4; ++nt)
    wb[nt] = W + (n0 + nt * 16 + lrow) * DDIM + lhalf * 16;

  BFrag faA, faB;
  BFrag fbA[4], fbB[4];

  // prologue: fill set A for k0 = 0
  load_afrag(faA, arow, 0, lhalf);
#pragma unroll
  for (int nt = 0; nt < 4; ++nt) load_bfrag(fbA[nt], wb[nt], 0);

#pragma unroll 1
  for (int k0 = 0; k0 < DDIM; k0 += 64) {
    // prefetch set B for k0+32 (DDIM is a multiple of 64, always in range)
    load_afrag(faB, arow, k0 + 32, lhalf);
#pragma unroll
    for (int nt = 0; nt < 4; ++nt) load_bfrag(fbB[nt], wb[nt], k0 + 32);

    // compute with set A
#pragma unroll
    for (int nt = 0; nt < 4; ++nt)
      acc[nt] = __builtin_amdgcn_wmma_f32_16x16x32_bf16(
          false, faA.v, false, fbA[nt].v, (short)0, acc[nt], false, false);

    // prefetch set A for k0+64 (wrap harmlessly on the last iteration)
    const int kn = (k0 + 64 < DDIM) ? k0 + 64 : 0;
    load_afrag(faA, arow, kn, lhalf);
#pragma unroll
    for (int nt = 0; nt < 4; ++nt) load_bfrag(fbA[nt], wb[nt], kn);

    // compute with set B
#pragma unroll
    for (int nt = 0; nt < 4; ++nt)
      acc[nt] = __builtin_amdgcn_wmma_f32_16x16x32_bf16(
          false, faB.v, false, fbB[nt].v, (short)0, acc[nt], false, false);
  }

#pragma unroll
  for (int nt = 0; nt < 4; ++nt) {
    const int col  = n0 + nt * 16 + lrow;
    const float bv = bias ? bias[col] : 0.0f;
#pragma unroll
    for (int r = 0; r < 8; ++r) {
      const int row = m0 + r + lhalf * 8;   // C layout: VGPR r -> M = r + 8*lhalf
      C[row * DDIM + col] = acc[nt][r] + bv;
    }
  }
}

// ---------------- q/k/v = x @ {wq,wk,wv}^T ---------------------------------
// 3 matrices * 16 M-tiles * 12 N64-tiles = 576 wave-tiles; 8 waves/block.
__global__ void qkv_gemm_kernel(const unsigned short* __restrict__ xb,
                                const unsigned short* __restrict__ wqb,
                                const unsigned short* __restrict__ wkb,
                                const unsigned short* __restrict__ wvb,
                                float* __restrict__ q, float* __restrict__ k,
                                float* __restrict__ v) {
  int wid = blockIdx.x * 8 + (threadIdx.x >> 5);
  int mat = wid / 192;
  int rem = wid % 192;
  int mt  = rem / 12;
  int nt  = rem % 12;
  const unsigned short* W = (mat == 0) ? wqb : (mat == 1) ? wkb : wvb;
  float* C                = (mat == 0) ? q   : (mat == 1) ? k   : v;
  gemm_wave_16x64(xb, W, C, nullptr, mt * 16, nt * 64);
}

// ---------------- kv[b] = sum_j k*v ; pre[b,:] = q * kv * s (bf16) ---------
__global__ void kv_outpre_kernel(const float* __restrict__ q,
                                 const float* __restrict__ k,
                                 const float* __restrict__ v,
                                 unsigned short* __restrict__ pre,
                                 float scale) {
  const int b = blockIdx.x;
  const int t = threadIdx.x;          // 256 threads
  __shared__ float red[256];
  float s = 0.0f;
  for (int j = t; j < DDIM; j += 256)
    s += k[b * DDIM + j] * v[b * DDIM + j];
  red[t] = s;
  __syncthreads();
  for (int off = 128; off > 0; off >>= 1) {
    if (t < off) red[t] += red[t + off];
    __syncthreads();
  }
  const float c = red[0] * scale;
  for (int j = t; j < DDIM; j += 256)
    pre[b * DDIM + j] = f2bf(q[b * DDIM + j] * c);
}

// ---------------- out = pre @ wo^T + bo ------------------------------------
__global__ void out_gemm_kernel(const unsigned short* __restrict__ preb,
                                const unsigned short* __restrict__ wob,
                                const float* __restrict__ bo,
                                float* __restrict__ out) {
  int wid = blockIdx.x * 8 + (threadIdx.x >> 5);
  int mt  = wid / 12;
  int nt  = wid % 12;
  gemm_wave_16x64(preb, wob, out, bo, mt * 16, nt * 64);
}

// ---------------- attn[b,i,j] = q[b,i]*k[b,j]*s  (streaming NT stores) -----
__global__ void attn_kernel(const float* __restrict__ q,
                            const float* __restrict__ k,
                            float* __restrict__ attn, float scale) {
  const int b = blockIdx.x;
  const int t = threadIdx.x;          // 192 threads, each owns 4 columns
  __shared__ float qrow[DDIM];
  reinterpret_cast<f32x4*>(qrow)[t] =
      reinterpret_cast<const f32x4*>(q + b * DDIM)[t];
  f32x4 kv = reinterpret_cast<const f32x4*>(k + b * DDIM)[t];
  kv *= scale;
  __syncthreads();
  float* base = attn + (size_t)b * DDIM * DDIM + t * 4;
#pragma unroll 4
  for (int i = 0; i < DDIM; ++i) {
    f32x4 o = qrow[i] * kv;
    __builtin_nontemporal_store(o, reinterpret_cast<f32x4*>(base + (size_t)i * DDIM));
  }
}

// ---------------------------------------------------------------------------
extern "C" void kernel_launch(void* const* d_in, const int* in_sizes, int n_in,
                              void* d_out, int out_size, void* d_ws, size_t ws_size,
                              hipStream_t stream) {
  const float* x  = (const float*)d_in[0];   // 256x768
  const float* wq = (const float*)d_in[1];   // 768x768
  const float* wk = (const float*)d_in[2];
  const float* wv = (const float*)d_in[3];
  const float* wo = (const float*)d_in[4];
  const float* bo = (const float*)d_in[5];   // 768

  float* out  = (float*)d_out;                      // 256*768
  float* attn = (float*)d_out + BATCH * DDIM;       // 256*768*768

  // workspace layout (bytes)
  char* ws = (char*)d_ws;
  unsigned short* xb   = (unsigned short*)(ws);                 // 256*768 bf16
  unsigned short* wqb  = (unsigned short*)(ws + 0x0060000);     // 768*768 bf16 each
  unsigned short* wkb  = (unsigned short*)(ws + 0x0180000);
  unsigned short* wvb  = (unsigned short*)(ws + 0x02A0000);
  unsigned short* wob  = (unsigned short*)(ws + 0x03C0000);
  float*          qf   = (float*)(ws + 0x04E0000);              // 256*768 f32 each
  float*          kf   = (float*)(ws + 0x05A0000);
  float*          vf   = (float*)(ws + 0x0660000);
  unsigned short* preb = (unsigned short*)(ws + 0x0720000);     // 256*768 bf16

  const float scale = 1.0f / sqrtf((float)DDIM);

  // 1) convert inputs to bf16
  {
    int nx4 = BATCH * DDIM / 4;          // 49152
    int nw4 = DDIM * DDIM / 4;           // 147456
    cvt_bf16_kernel<<<(nx4 + 255) / 256, 256, 0, stream>>>(x,  xb,  nx4);
    cvt_bf16_kernel<<<(nw4 + 255) / 256, 256, 0, stream>>>(wq, wqb, nw4);
    cvt_bf16_kernel<<<(nw4 + 255) / 256, 256, 0, stream>>>(wk, wkb, nw4);
    cvt_bf16_kernel<<<(nw4 + 255) / 256, 256, 0, stream>>>(wv, wvb, nw4);
    cvt_bf16_kernel<<<(nw4 + 255) / 256, 256, 0, stream>>>(wo, wob, nw4);
  }

  // 2) q/k/v GEMMs: 576 wave-tiles / 8 waves per block
  qkv_gemm_kernel<<<72, 256, 0, stream>>>(xb, wqb, wkb, wvb, qf, kf, vf);

  // 3) kv contraction + pre-activation for the output projection
  kv_outpre_kernel<<<BATCH, 256, 0, stream>>>(qf, kf, vf, preb, scale);

  // 4) out = pre @ wo^T + bo : 192 wave-tiles / 8 per block
  out_gemm_kernel<<<24, 256, 0, stream>>>(preb, wob, bo, out);

  // 5) attn scores: 604 MB streaming store — the real roofline term
  attn_kernel<<<BATCH, 192, 0, stream>>>(qf, kf, attn, scale);
}
